// TemporalGAT_9345848836369
// MI455X (gfx1250) — matrix operand order repaired and checked by reading, not compiled, not run on hardware
//
#include <hip/hip_runtime.h>
#include <hip/hip_bf16.h>
#include <math.h>

#define NEG_SLOPE 0.2f
#define EPSV 1e-5f

typedef __attribute__((ext_vector_type(16))) __bf16 v16bf;
typedef __attribute__((ext_vector_type(8)))  float  v8f;

static __device__ __forceinline__ unsigned fkey(float f) {
    unsigned u = __float_as_uint(f);
    return (u & 0x80000000u) ? ~u : (u | 0x80000000u);
}
static __device__ __forceinline__ float funkey(unsigned u) {
    return (u & 0x80000000u) ? __uint_as_float(u ^ 0x80000000u) : __uint_as_float(~u);
}
static __device__ __forceinline__ float leaky(float x) { return x > 0.f ? x : NEG_SLOPE * x; }

// ---------------- utility kernels ----------------

__global__ void k_zero(float* __restrict__ p, int n) {
    int i = blockIdx.x * blockDim.x + threadIdx.x;
    if (i < n) p[i] = 0.f;
}

// W[K,Nn] f32 row-major -> Wt[Nn,Kpad] bf16 (transposed, K contiguous, zero-padded to Kpad)
__global__ void k_cvt_wt(const float* __restrict__ W, __bf16* __restrict__ Wt,
                         int K, int Kpad, int Nn) {
    int i = blockIdx.x * blockDim.x + threadIdx.x;
    if (i >= Nn * Kpad) return;
    int n = i / Kpad, k = i - n * Kpad;
    Wt[i] = (k < K) ? (__bf16)W[(size_t)k * Nn + n] : (__bf16)0.0f;
}

// Fourier time encoding of each edge, scatter-add onto source nodes
__global__ void k_time_scatter(const float* __restrict__ et, const long long* __restrict__ src,
                               const float* __restrict__ tw, const float* __restrict__ tb,
                               float* __restrict__ tsum, float* __restrict__ cnt, int E) {
    int i = blockIdx.x * blockDim.x + threadIdx.x;
    if (i >= E * 16) return;
    int e = i >> 4, j = i & 15;
    float tp = et[e] * tw[j] + tb[j];
    float v = ((j & 1) == 0) ? sinf(tp) : cosf(tp);
    long long s = src[e];
    atomicAdd(&tsum[s * 16 + j], v);
    if (j == 0) atomicAdd(&cnt[s], 1.f);
}

// xc[n,0:128]=x ; xc[n,128:144]=tsum/(cnt+1) ; xc[n,144:160]=0   (K padded to 160)
__global__ void k_concat(const float* __restrict__ x, const float* __restrict__ tsum,
                         const float* __restrict__ cnt, float* __restrict__ xc, int N) {
    int i = blockIdx.x * blockDim.x + threadIdx.x;
    if (i >= N * 160) return;
    int n = i / 160, j = i - n * 160;
    float v;
    if (j < 128)      v = x[(size_t)n * 128 + j];
    else if (j < 144) v = tsum[n * 16 + (j - 128)] / (cnt[n] + 1.f);
    else              v = 0.f;
    xc[i] = v;
}

// ---------------- WMMA bf16 GEMM ----------------
// C[M,Nn] = A[M,K](f32, row-major, K%32==0) @ B ; Bt = B^T bf16 [Nn,K].
// One wave computes a 16 x (16*NW) tile: one A fragment feeds NW WMMAs per K-step.
template <int NW>
__global__ void k_gemm_wmma(const float* __restrict__ A, const __bf16* __restrict__ Bt,
                            const float* __restrict__ bias, float* __restrict__ C,
                            int M, int K, int Nn) {
    int gw   = (int)(((size_t)blockIdx.x * blockDim.x + threadIdx.x) >> 5);
    int lane = threadIdx.x & 31;
    int tilesN = Nn / (16 * NW);
    int tM = gw / tilesN;
    int tN = gw - tM * tilesN;
    if (tM * 16 >= M) return;
    int hl = lane >> 4;      // K-half selector
    int lm = lane & 15;      // M (for A) / N (for B,C) within 16-subtile
    const float* ap = A + (size_t)(tM * 16 + lm) * K;
    const __bf16* bp[NW];
#pragma unroll
    for (int j = 0; j < NW; ++j)
        bp[j] = Bt + (size_t)(tN * NW * 16 + j * 16 + lm) * K;

    v8f acc[NW];
#pragma unroll
    for (int j = 0; j < NW; ++j) acc[j] = {};

    for (int kb = 0; kb < K; kb += 32) {
        // A fragment: elem i -> k = kb + hl*16 + i; two contiguous 8-float runs
        v16bf a;
#pragma unroll
        for (int c = 0; c < 2; ++c) {
            int k0 = kb + (hl << 4) + c * 8;
            float4 f0 = *(const float4*)(ap + k0);
            float4 f1 = *(const float4*)(ap + k0 + 4);
            a[c * 8 + 0] = (__bf16)f0.x; a[c * 8 + 1] = (__bf16)f0.y;
            a[c * 8 + 2] = (__bf16)f0.z; a[c * 8 + 3] = (__bf16)f0.w;
            a[c * 8 + 4] = (__bf16)f1.x; a[c * 8 + 5] = (__bf16)f1.y;
            a[c * 8 + 6] = (__bf16)f1.z; a[c * 8 + 7] = (__bf16)f1.w;
        }
        // B fragment j: elem i -> k = kb + hl*16 + i ; contiguous 16 bf16 (32B)
        int kB = kb + (hl << 4);
#pragma unroll
        for (int j = 0; j < NW; ++j) {
            v16bf b = *(const v16bf*)(bp[j] + kB);
            acc[j] = __builtin_amdgcn_wmma_f32_16x16x32_bf16(false, a, false, b,
                                                             (short)0, acc[j], false, false);
        }
    }
    // D: elem r -> m = tM*16 + hl*8 + r, n = subtile col + lm
#pragma unroll
    for (int j = 0; j < NW; ++j) {
        int n = tN * NW * 16 + j * 16 + lm;
        float bv = bias ? bias[n] : 0.f;
#pragma unroll
        for (int r = 0; r < 8; ++r) {
            int m = tM * 16 + hl * 8 + r;
            C[(size_t)m * Nn + n] = acc[j][r] + bv;
        }
    }
}

// ---------------- attention kernels (generic over H heads, C channels/head) ----------------

__global__ void k_alpha(const float* __restrict__ h, const float* __restrict__ asrc,
                        const float* __restrict__ adst, float* __restrict__ als,
                        float* __restrict__ ald, unsigned* __restrict__ mkey,
                        int N, int H, int C) {
    int i = blockIdx.x * blockDim.x + threadIdx.x;
    if (i >= N * H) return;
    int n = i / H, hd = i - n * H;
    const float* hp = h + (size_t)n * H * C + hd * C;
    float s = 0.f, d = 0.f;
    for (int c = 0; c < C; ++c) { float v = hp[c]; s += v * asrc[hd * C + c]; d += v * adst[hd * C + c]; }
    als[i] = s; ald[i] = d;
    mkey[i] = fkey(leaky(s + d));   // self-loop edge seeds the segment max
}

__global__ void k_edge_max(const long long* __restrict__ src, const long long* __restrict__ dst,
                           const float* __restrict__ als, const float* __restrict__ ald,
                           unsigned* __restrict__ mkey, int E, int H) {
    int i = blockIdx.x * blockDim.x + threadIdx.x;
    if (i >= E * H) return;
    int e = i / H, hd = i - e * H;
    float v = leaky(als[src[e] * H + hd] + ald[dst[e] * H + hd]);
    atomicMax(&mkey[dst[e] * H + hd], fkey(v));
}

__global__ void k_self_init(const float* __restrict__ als, const float* __restrict__ ald,
                            const unsigned* __restrict__ mkey, float* __restrict__ denom,
                            float* __restrict__ selfex, int NH) {
    int i = blockIdx.x * blockDim.x + threadIdx.x;
    if (i >= NH) return;
    float ex = expf(leaky(als[i] + ald[i]) - funkey(mkey[i]));
    denom[i] = ex; selfex[i] = ex;
}

__global__ void k_agg_self(const float* __restrict__ h, const float* __restrict__ selfex,
                           float* __restrict__ agg, int N, int H, int C) {
    int i = blockIdx.x * blockDim.x + threadIdx.x;
    if (i >= N * H * C) return;
    int n = i / (H * C); int r = i - n * H * C; int hd = r / C;
    agg[i] = h[i] * selfex[n * H + hd];
}

__global__ void k_edge_exp(const long long* __restrict__ src, const long long* __restrict__ dst,
                           const float* __restrict__ als, const float* __restrict__ ald,
                           const unsigned* __restrict__ mkey, float* __restrict__ exbuf,
                           float* __restrict__ denom, int E, int H) {
    int i = blockIdx.x * blockDim.x + threadIdx.x;
    if (i >= E * H) return;
    int e = i / H, hd = i - e * H;
    long long d = dst[e];
    float v = leaky(als[src[e] * H + hd] + ald[d * H + hd]);
    float ex = expf(v - funkey(mkey[d * H + hd]));
    exbuf[i] = ex;
    atomicAdd(&denom[d * H + hd], ex);
}

__global__ void k_edge_agg(const long long* __restrict__ src, const long long* __restrict__ dst,
                           const float* __restrict__ h, const float* __restrict__ exbuf,
                           float* __restrict__ agg, int E, int H, int C) {
    int i = blockIdx.x * blockDim.x + threadIdx.x;
    int HC = H * C;
    if (i >= E * HC) return;
    int e = i / HC; int r = i - e * HC; int hd = r / C;
    atomicAdd(&agg[dst[e] * (size_t)HC + r],
              h[src[e] * (size_t)HC + r] * exbuf[(size_t)e * H + hd]);
}

// layer1 epilogue: /denom + bias -> ELU -> LayerNorm, in place (256 ch, 4 heads)
__global__ void k_post1(float* __restrict__ agg, const float* __restrict__ denom,
                        const float* __restrict__ bias, const float* __restrict__ g,
                        const float* __restrict__ be, int N) {
    int n = blockIdx.x * blockDim.x + threadIdx.x;
    if (n >= N) return;
    float* row = agg + (size_t)n * 256;
    const float* dn = denom + n * 4;
    float sum = 0.f, sq = 0.f;
    for (int c = 0; c < 256; ++c) {
        float v = row[c] / dn[c >> 6] + bias[c];
        v = v > 0.f ? v : expm1f(v);
        sum += v; sq += v * v;
    }
    float mu = sum * (1.f / 256.f);
    float var = sq * (1.f / 256.f) - mu * mu;
    float rs = rsqrtf(var + EPSV);
    for (int c = 0; c < 256; ++c) {
        float v = row[c] / dn[c >> 6] + bias[c];
        v = v > 0.f ? v : expm1f(v);
        row[c] = (v - mu) * rs * g[c] + be[c];
    }
}

// layer2 epilogue: /denom + bias -> LayerNorm -> d_out (32 ch, 1 head)
__global__ void k_post2(const float* __restrict__ agg, const float* __restrict__ denom,
                        const float* __restrict__ bias, const float* __restrict__ g,
                        const float* __restrict__ be, float* __restrict__ out, int N) {
    int n = blockIdx.x * blockDim.x + threadIdx.x;
    if (n >= N) return;
    const float* row = agg + (size_t)n * 32;
    float dn = denom[n];
    float vals[32];
    float sum = 0.f, sq = 0.f;
    for (int c = 0; c < 32; ++c) {
        float v = row[c] / dn + bias[c];
        vals[c] = v; sum += v; sq += v * v;
    }
    float mu = sum * (1.f / 32.f);
    float var = sq * (1.f / 32.f) - mu * mu;
    float rs = rsqrtf(var + EPSV);
    for (int c = 0; c < 32; ++c)
        out[(size_t)n * 32 + c] = (vals[c] - mu) * rs * g[c] + be[c];
}

// ---------------- host launcher ----------------

static inline int cdiv(long long a, int b) { return (int)((a + b - 1) / b); }

extern "C" void kernel_launch(void* const* d_in, const int* in_sizes, int n_in,
                              void* d_out, int out_size, void* d_ws, size_t ws_size,
                              hipStream_t stream) {
    (void)n_in; (void)out_size; (void)ws_size;
    const float*     x      = (const float*)d_in[0];
    const long long* ei     = (const long long*)d_in[1];
    const float*     et     = (const float*)d_in[2];
    const float*     tw     = (const float*)d_in[3];
    const float*     tb     = (const float*)d_in[4];
    const float*     proj_w = (const float*)d_in[5];
    const float*     proj_b = (const float*)d_in[6];
    const float*     W1     = (const float*)d_in[7];
    const float*     a_src1 = (const float*)d_in[8];
    const float*     a_dst1 = (const float*)d_in[9];
    const float*     b1     = (const float*)d_in[10];
    const float*     g1     = (const float*)d_in[11];
    const float*     be1    = (const float*)d_in[12];
    const float*     W2     = (const float*)d_in[13];
    const float*     a_src2 = (const float*)d_in[14];
    const float*     a_dst2 = (const float*)d_in[15];
    const float*     b2     = (const float*)d_in[16];
    const float*     g2     = (const float*)d_in[17];
    const float*     be2    = (const float*)d_in[18];

    const int N = in_sizes[0] / 128;
    const int E = in_sizes[2];
    const long long* src = ei;
    const long long* dst = ei + E;

    // scratch carve (256B aligned slabs)
    char* base = (char*)d_ws;
    size_t off = 0;
    auto carve = [&](size_t bytes) -> char* {
        char* p = base + off;
        off += (bytes + 255) & ~(size_t)255;
        return p;
    };
    float*    tsum   = (float*)carve((size_t)N * 16 * 4);
    float*    cnt    = (float*)carve((size_t)N * 4);
    float*    xc     = (float*)carve((size_t)N * 160 * 4);      // K padded 144->160
    __bf16*   wtP    = (__bf16*)carve((size_t)128 * 160 * 2);   // [Nn=128, Kpad=160]
    __bf16*   wt1    = (__bf16*)carve((size_t)256 * 128 * 2);   // [Nn=256, K=128]
    __bf16*   wt2    = (__bf16*)carve((size_t)32 * 256 * 2);    // [Nn=32,  K=256]
    float*    xin    = (float*)carve((size_t)N * 128 * 4);
    float*    h1     = (float*)carve((size_t)N * 256 * 4);
    float*    als1   = (float*)carve((size_t)N * 4 * 4);
    float*    ald1   = (float*)carve((size_t)N * 4 * 4);
    unsigned* m1     = (unsigned*)carve((size_t)N * 4 * 4);
    float*    den1   = (float*)carve((size_t)N * 4 * 4);
    float*    sex1   = (float*)carve((size_t)N * 4 * 4);
    float*    ex1    = (float*)carve((size_t)E * 4 * 4);
    float*    agg1   = (float*)carve((size_t)N * 256 * 4);      // becomes h1n in place
    float*    h2     = (float*)carve((size_t)N * 32 * 4);
    float*    als2   = (float*)carve((size_t)N * 4);
    float*    ald2   = (float*)carve((size_t)N * 4);
    unsigned* m2     = (unsigned*)carve((size_t)N * 4);
    float*    den2   = (float*)carve((size_t)N * 4);
    float*    sex2   = (float*)carve((size_t)N * 4);
    float*    ex2    = (float*)carve((size_t)E * 4);
    float*    agg2   = (float*)carve((size_t)N * 32 * 4);

    const int T = 256;

    // stage 0: weights -> bf16 transposed (K-padded); time-encode scatter-mean; concat input
    k_zero<<<cdiv((long long)N * 16, T), T, 0, stream>>>(tsum, N * 16);
    k_zero<<<cdiv(N, T), T, 0, stream>>>(cnt, N);
    k_cvt_wt<<<cdiv(128 * 160, T), T, 0, stream>>>(proj_w, wtP, 144, 160, 128);
    k_cvt_wt<<<cdiv(256 * 128, T), T, 0, stream>>>(W1, wt1, 128, 128, 256);
    k_cvt_wt<<<cdiv(32 * 256, T), T, 0, stream>>>(W2, wt2, 256, 256, 32);
    k_time_scatter<<<cdiv((long long)E * 16, T), T, 0, stream>>>(et, src, tw, tb, tsum, cnt, E);
    k_concat<<<cdiv((long long)N * 160, T), T, 0, stream>>>(x, tsum, cnt, xc, N);

    // projection GEMM: [N,160pad] @ -> [N,128] (+bias), 16x64 tile/wave
    {
        long long waves = (long long)(N / 16) * (128 / 64);
        k_gemm_wmma<4><<<cdiv(waves * 32, T), T, 0, stream>>>(xc, wtP, proj_b, xin, N, 160, 128);
    }
    // layer-1 GEMM: [N,128] @ -> [N,256], 16x64 tile/wave
    {
        long long waves = (long long)(N / 16) * (256 / 64);
        k_gemm_wmma<4><<<cdiv(waves * 32, T), T, 0, stream>>>(xin, wt1, nullptr, h1, N, 128, 256);
    }
    // layer-1 attention (4 heads, 64 ch)
    k_alpha<<<cdiv((long long)N * 4, T), T, 0, stream>>>(h1, a_src1, a_dst1, als1, ald1, m1, N, 4, 64);
    k_edge_max<<<cdiv((long long)E * 4, T), T, 0, stream>>>(src, dst, als1, ald1, m1, E, 4);
    k_self_init<<<cdiv((long long)N * 4, T), T, 0, stream>>>(als1, ald1, m1, den1, sex1, N * 4);
    k_agg_self<<<cdiv((long long)N * 256, T), T, 0, stream>>>(h1, sex1, agg1, N, 4, 64);
    k_edge_exp<<<cdiv((long long)E * 4, T), T, 0, stream>>>(src, dst, als1, ald1, m1, ex1, den1, E, 4);
    k_edge_agg<<<cdiv((long long)E * 256, T), T, 0, stream>>>(src, dst, h1, ex1, agg1, E, 4, 64);
    k_post1<<<cdiv(N, T), T, 0, stream>>>(agg1, den1, b1, g1, be1, N);

    // layer-2 GEMM: [N,256] @ -> [N,32], 16x32 tile/wave
    {
        long long waves = (long long)(N / 16) * (32 / 32);
        k_gemm_wmma<2><<<cdiv(waves * 32, T), T, 0, stream>>>(agg1, wt2, nullptr, h2, N, 256, 32);
    }
    // layer-2 attention (1 head, 32 ch)
    k_alpha<<<cdiv(N, T), T, 0, stream>>>(h2, a_src2, a_dst2, als2, ald2, m2, N, 1, 32);
    k_edge_max<<<cdiv(E, T), T, 0, stream>>>(src, dst, als2, ald2, m2, E, 1);
    k_self_init<<<cdiv(N, T), T, 0, stream>>>(als2, ald2, m2, den2, sex2, N);
    k_agg_self<<<cdiv((long long)N * 32, T), T, 0, stream>>>(h2, sex2, agg2, N, 1, 32);
    k_edge_exp<<<cdiv(E, T), T, 0, stream>>>(src, dst, als2, ald2, m2, ex2, den2, E, 1);
    k_edge_agg<<<cdiv((long long)E * 32, T), T, 0, stream>>>(src, dst, h2, ex2, agg2, E, 1, 32);
    k_post2<<<cdiv(N, T), T, 0, stream>>>(agg2, den2, b2, g2, be2, (float*)d_out, N);
}
